// MoEContradictionClassifier_16149077033522
// MI455X (gfx1250) — compile-verified, matrix operand order checked by reading, not run on hardware
//
#include <hip/hip_runtime.h>
#include <hip/hip_bf16.h>
#include <math.h>

// ---------------------------------------------------------------------------
// MoE contradiction classifier for MI455X (gfx1250, wave32, WMMA).
// All matmuls run on v_wmma_f32_16x16x32_bf16 (fp32 accumulate).
// ---------------------------------------------------------------------------

typedef __bf16 b16;
typedef __attribute__((ext_vector_type(16))) __bf16 v16b;
typedef __attribute__((ext_vector_type(8)))  __bf16 v8b;
typedef __attribute__((ext_vector_type(8)))  float  v8f;

#define TOK    8192
#define BATCH  64
#define SEQ    128
#define HID    768
#define FFD    3072
#define NHEAD  12
#define HDIM   64
#define NEXP   4
#define NLAY   2

enum { ACT_NONE = 0, ACT_GELU = 1, ACT_RELU = 2 };

__device__ __forceinline__ float gelu_f(float x) {
  float x3 = x * x * x;
  return 0.5f * x * (1.0f + tanhf(0.7978845608028654f * (x + 0.044715f * x3)));
}

__device__ __forceinline__ v8f zero8() {
  v8f z = {0.f, 0.f, 0.f, 0.f, 0.f, 0.f, 0.f, 0.f};
  return z;
}

union frag_u { v16b v; v8b h[2]; };

// Load one 16x32 bf16 WMMA operand tile from LDS.
// ISA 16-bit A layout (wave32): lanes 0-15 hold row M=lane with K = {0..7,16..23},
// lanes 16-31 hold row M=lane-16 with K = {8..15,24..31}. Two 16B chunks per lane.
// B operand uses the mirrored per-column layout, so the same gather works when
// the B tile is stored column-major (one column's K-values contiguous).
__device__ __forceinline__ v16b load_frag(const b16* __restrict__ base, int stride) {
  int lane = threadIdx.x & 31;
  int lh = lane >> 4;
  int r  = lane & 15;
  const b16* p = base + r * stride + lh * 8;
  frag_u u;
  u.h[0] = *(const v8b*)(p);
  u.h[1] = *(const v8b*)(p + 16);
  return u.v;
}

__device__ __forceinline__ v8f wmma_bf16(v16b a, v16b b, v8f c) {
  return __builtin_amdgcn_wmma_f32_16x16x32_bf16(false, a, false, b, (short)0, c,
                                                 false, false);
}

// ---------------------------------------------------------------------------
// Generic GEMM: C[M,N] = act(A[M,K](lda) @ B[K,N] + bias)
// fp32 in/out, bf16 WMMA inside. Block tile 128x128, 8 waves, wave tile 32x64.
// K must be a multiple of 32 (true for all call sites: 768, 3072, 256).
// ---------------------------------------------------------------------------
#define GT_M 128
#define GT_N 128
#define GT_K 32

__global__ __launch_bounds__(256)
void gemm_kernel(const float* __restrict__ A, int lda,
                 const float* __restrict__ B,
                 const float* __restrict__ bias,
                 float* __restrict__ C,
                 int M, int N, int K, int act) {
  __shared__ alignas(16) b16 Asm[GT_M * GT_K];          // [row][k]
  __shared__ alignas(16) b16 Bsm[GT_N * GT_K];          // [col][k] (column-major)

  const int m0 = blockIdx.y * GT_M;
  const int n0 = blockIdx.x * GT_N;
  const int tid  = threadIdx.x;
  const int wave = tid >> 5;
  const int lane = tid & 31;
  const int wm = (wave >> 1) * 32;   // wave row offset: 0,32,64,96
  const int wn = (wave & 1) * 64;    // wave col offset: 0,64

  v8f acc[2][4];
#pragma unroll
  for (int mi = 0; mi < 2; ++mi)
#pragma unroll
    for (int ni = 0; ni < 4; ++ni) acc[mi][ni] = zero8();

  for (int k0 = 0; k0 < K; k0 += GT_K) {
    // ---- stage A tile: 128 rows x 32 k, fp32 -> bf16 ----
    {
      int row = tid >> 1;
      int kx  = (tid & 1) * 16;
      int gr  = m0 + row;
      b16* dst = Asm + row * GT_K + kx;
      if (gr < M) {
        const float* src = A + (size_t)gr * lda + k0 + kx;
        if (k0 + GT_K < K) __builtin_prefetch(src + GT_K, 0, 0);
#pragma unroll
        for (int i = 0; i < 16; i += 4) {
          float4 f = *(const float4*)(src + i);
          dst[i + 0] = (b16)f.x; dst[i + 1] = (b16)f.y;
          dst[i + 2] = (b16)f.z; dst[i + 3] = (b16)f.w;
        }
      } else {
#pragma unroll
        for (int i = 0; i < 16; ++i) dst[i] = (b16)0.0f;
      }
    }
    // ---- stage B tile: 32 k-rows x 128 cols, transpose into [col][k] ----
    {
      int kr = tid >> 3;
      int nx = (tid & 7) * 16;
      const float* src = B + (size_t)(k0 + kr) * N + n0 + nx;
      if (k0 + GT_K < K) __builtin_prefetch(src + (size_t)GT_K * N, 0, 0);
#pragma unroll
      for (int i = 0; i < 16; ++i) {
        int gn = n0 + nx + i;
        float v = (gn < N) ? src[i] : 0.0f;
        Bsm[(nx + i) * GT_K + kr] = (b16)v;
      }
    }
    __syncthreads();

    v16b af[2], bfq[4];
#pragma unroll
    for (int mi = 0; mi < 2; ++mi)
      af[mi] = load_frag(Asm + (wm + 16 * mi) * GT_K, GT_K);
#pragma unroll
    for (int ni = 0; ni < 4; ++ni)
      bfq[ni] = load_frag(Bsm + (wn + 16 * ni) * GT_K, GT_K);
#pragma unroll
    for (int mi = 0; mi < 2; ++mi)
#pragma unroll
      for (int ni = 0; ni < 4; ++ni)
        acc[mi][ni] = wmma_bf16(af[mi], bfq[ni], acc[mi][ni]);
    __syncthreads();
  }

  // ---- epilogue: D layout -> row = i + 8*half, col = lane&15 ----
  const int lh = lane >> 4;
  const int r  = lane & 15;
#pragma unroll
  for (int mi = 0; mi < 2; ++mi) {
#pragma unroll
    for (int ni = 0; ni < 4; ++ni) {
#pragma unroll
      for (int i = 0; i < 8; ++i) {
        int row = m0 + wm + 16 * mi + i + 8 * lh;
        int col = n0 + wn + 16 * ni + r;
        if (row < M && col < N) {
          float v = acc[mi][ni][i];
          if (bias) v += bias[col];
          if (act == ACT_GELU)      v = gelu_f(v);
          else if (act == ACT_RELU) v = fmaxf(v, 0.0f);
          C[(size_t)row * N + col] = v;
        }
      }
    }
  }
}

// ---------------------------------------------------------------------------
// Fused attention: one block per (batch, head). S=128, D=64.
// scores = (Q @ K^T)*scale + mask_bias ; P = softmax(scores) ; ctx = P @ V
// Scores stay in registers (64 VGPRs/wave); softmax via 16-lane shfl reductions
// (matches the WMMA D layout where one row lives in one half-wave).
// ---------------------------------------------------------------------------
__global__ __launch_bounds__(256)
void attn_kernel(const float* __restrict__ qkv, const int* __restrict__ mask,
                 float* __restrict__ ctx) {
  const int b = blockIdx.x / NHEAD;
  const int h = blockIdx.x % NHEAD;

  __shared__ alignas(16) union {
    struct { b16 q[SEQ * HDIM]; b16 k[SEQ * HDIM]; } s;   // 32 KB
    b16 p[SEQ * SEQ];                                     // 32 KB (reuses q/k)
  } u;
  __shared__ alignas(16) b16 vt[HDIM * SEQ];              // V^T: [d][s], 16 KB
  __shared__ float biasr[SEQ];

  const int tid = threadIdx.x;
  // ---- stage Q, K, V^T as bf16 ----
  {
    int s  = tid >> 1;
    int d0 = (tid & 1) * 32;
    const float* base = qkv + ((size_t)(b * SEQ + s)) * (3 * HID) + h * HDIM + d0;
#pragma unroll
    for (int i = 0; i < 32; i += 4) {
      float4 fq = *(const float4*)(base + i);
      float4 fk = *(const float4*)(base + HID + i);
      float4 fv = *(const float4*)(base + 2 * HID + i);
      u.s.q[s * HDIM + d0 + i + 0] = (b16)fq.x; u.s.q[s * HDIM + d0 + i + 1] = (b16)fq.y;
      u.s.q[s * HDIM + d0 + i + 2] = (b16)fq.z; u.s.q[s * HDIM + d0 + i + 3] = (b16)fq.w;
      u.s.k[s * HDIM + d0 + i + 0] = (b16)fk.x; u.s.k[s * HDIM + d0 + i + 1] = (b16)fk.y;
      u.s.k[s * HDIM + d0 + i + 2] = (b16)fk.z; u.s.k[s * HDIM + d0 + i + 3] = (b16)fk.w;
      vt[(d0 + i + 0) * SEQ + s] = (b16)fv.x;
      vt[(d0 + i + 1) * SEQ + s] = (b16)fv.y;
      vt[(d0 + i + 2) * SEQ + s] = (b16)fv.z;
      vt[(d0 + i + 3) * SEQ + s] = (b16)fv.w;
    }
  }
  if (tid < SEQ) biasr[tid] = (1.0f - (float)mask[b * SEQ + tid]) * -1e9f;
  __syncthreads();

  const int wave = tid >> 5;
  const int lane = tid & 31;
  const int lh = lane >> 4;
  const int r  = lane & 15;
  const int mrow = wave * 16;     // each wave owns 16 query rows

  // ---- phase 1: scores = Q @ K^T (M=16/wave, N=128, K=64) ----
  v8f acc[8];
#pragma unroll
  for (int nt = 0; nt < 8; ++nt) acc[nt] = zero8();
#pragma unroll
  for (int kk = 0; kk < HDIM; kk += 32) {
    v16b a = load_frag(u.s.q + mrow * HDIM + kk, HDIM);
#pragma unroll
    for (int nt = 0; nt < 8; ++nt) {
      v16b kb = load_frag(u.s.k + nt * 16 * HDIM + kk, HDIM);
      acc[nt] = wmma_bf16(a, kb, acc[nt]);
    }
  }

  const float scale = 0.125f;  // 1/sqrt(64)
#pragma unroll
  for (int nt = 0; nt < 8; ++nt)
#pragma unroll
    for (int i = 0; i < 8; ++i)
      acc[nt][i] = acc[nt][i] * scale + biasr[nt * 16 + r];

  // ---- register softmax: row m = mrow + i + 8*lh spans 8 tiles x 16 lanes ----
  float rinv[8];
#pragma unroll
  for (int i = 0; i < 8; ++i) {
    float m = acc[0][i];
#pragma unroll
    for (int nt = 1; nt < 8; ++nt) m = fmaxf(m, acc[nt][i]);
#pragma unroll
    for (int off = 8; off >= 1; off >>= 1) m = fmaxf(m, __shfl_xor(m, off, 32));
    float s = 0.0f;
#pragma unroll
    for (int nt = 0; nt < 8; ++nt) {
      acc[nt][i] = __expf(acc[nt][i] - m);
      s += acc[nt][i];
    }
#pragma unroll
    for (int off = 8; off >= 1; off >>= 1) s += __shfl_xor(s, off, 32);
    rinv[i] = 1.0f / s;
  }

  __syncthreads();   // all waves done reading q/k before P overwrites them
#pragma unroll
  for (int nt = 0; nt < 8; ++nt)
#pragma unroll
    for (int i = 0; i < 8; ++i)
      u.p[(mrow + i + 8 * lh) * SEQ + nt * 16 + r] = (b16)(acc[nt][i] * rinv[i]);
  __syncthreads();

  // ---- phase 2: ctx = P @ V (M=16/wave, N=64, K=128) ----
  v8f acc2[4];
#pragma unroll
  for (int nt = 0; nt < 4; ++nt) acc2[nt] = zero8();
#pragma unroll
  for (int kk = 0; kk < SEQ; kk += 32) {
    v16b a = load_frag(u.p + mrow * SEQ + kk, SEQ);
#pragma unroll
    for (int nt = 0; nt < 4; ++nt) {
      v16b vb = load_frag(vt + nt * 16 * SEQ + kk, SEQ);
      acc2[nt] = wmma_bf16(a, vb, acc2[nt]);
    }
  }
#pragma unroll
  for (int nt = 0; nt < 4; ++nt)
#pragma unroll
    for (int i = 0; i < 8; ++i) {
      int s = mrow + i + 8 * lh;
      int d = nt * 16 + r;
      ctx[((size_t)(b * SEQ + s)) * HID + h * HDIM + d] = acc2[nt][i];
    }
}

// ---------------------------------------------------------------------------
// out[row] = act( LN(in[row] (+ res[row])) * sc + bi )   one block per row
// ---------------------------------------------------------------------------
__global__ __launch_bounds__(256)
void ln_kernel(const float* __restrict__ in, const float* __restrict__ res,
               float* __restrict__ out, const float* __restrict__ sc,
               const float* __restrict__ bi, int H, int act) {
  const size_t row = blockIdx.x;
  const float* ir = in + row * H;
  const float* rr = res ? res + row * H : nullptr;
  float* orow = out + row * H;

  float lsum = 0.0f, lsq = 0.0f;
  for (int i = threadIdx.x; i < H; i += 256) {
    float v = ir[i] + (rr ? rr[i] : 0.0f);
    orow[i] = v;
    lsum += v;
    lsq  += v * v;
  }
#pragma unroll
  for (int off = 16; off >= 1; off >>= 1) {
    lsum += __shfl_xor(lsum, off, 32);
    lsq  += __shfl_xor(lsq,  off, 32);
  }
  __shared__ float s1[8], s2[8];
  __shared__ float smean, sinv;
  int wv = threadIdx.x >> 5;
  if ((threadIdx.x & 31) == 0) { s1[wv] = lsum; s2[wv] = lsq; }
  __syncthreads();
  if (threadIdx.x == 0) {
    float a = 0.0f, q = 0.0f;
#pragma unroll
    for (int i = 0; i < 8; ++i) { a += s1[i]; q += s2[i]; }
    float mean = a / (float)H;
    float var  = q / (float)H - mean * mean;
    smean = mean;
    sinv  = rsqrtf(var + 1e-5f);
  }
  __syncthreads();
  float mean = smean, inv = sinv;
  for (int i = threadIdx.x; i < H; i += 256) {
    float v = (orow[i] - mean) * inv * sc[i] + bi[i];
    if (act == ACT_GELU)      v = gelu_f(v);
    else if (act == ACT_RELU) v = fmaxf(v, 0.0f);
    orow[i] = v;
  }
}

// ---------------------------------------------------------------------------
// small helper kernels
// ---------------------------------------------------------------------------
__global__ __launch_bounds__(256)
void embed_kernel(const int* __restrict__ ids, const float* __restrict__ embed,
                  float* __restrict__ x0) {
  size_t t = blockIdx.x;
  size_t id = (size_t)ids[t];
  for (int i = threadIdx.x; i < HID; i += 256)
    x0[t * HID + i] = embed[id * HID + i];
}

__global__ __launch_bounds__(256)
void cls_copy_kernel(const float* __restrict__ x, float* __restrict__ dst) {
  size_t b = blockIdx.x;
  for (int i = threadIdx.x; i < HID; i += 256)
    dst[b * HID + i] = x[b * SEQ * HID + i];
}

// gating softmax + top-2 combine weights (64 rows x 4 experts)
__global__ void gate_head_kernel(const float* __restrict__ logits,
                                 float* __restrict__ probs_out,
                                 float* __restrict__ w) {
  int b = threadIdx.x;
  if (b >= BATCH) return;
  float v[NEXP], p[NEXP];
  float m = -1e30f;
#pragma unroll
  for (int e = 0; e < NEXP; ++e) { v[e] = logits[b * NEXP + e]; m = fmaxf(m, v[e]); }
  float s = 0.0f;
#pragma unroll
  for (int e = 0; e < NEXP; ++e) { p[e] = __expf(v[e] - m); s += p[e]; }
  float inv = 1.0f / s;
#pragma unroll
  for (int e = 0; e < NEXP; ++e) { p[e] *= inv; probs_out[b * NEXP + e] = p[e]; }
  int i1 = 0;
#pragma unroll
  for (int e = 1; e < NEXP; ++e) if (p[e] > p[i1]) i1 = e;
  int i2 = -1;
#pragma unroll
  for (int e = 0; e < NEXP; ++e)
    if (e != i1 && (i2 < 0 || p[e] > p[i2])) i2 = e;
#pragma unroll
  for (int e = 0; e < NEXP; ++e)
    w[b * NEXP + e] = (e == i1 || e == i2) ? p[e] : 0.0f;
}

__global__ __launch_bounds__(256)
void combine_kernel(const float* __restrict__ w, const float* __restrict__ ecls,
                    float* __restrict__ comb) {
  int b = blockIdx.x;
  for (int i = threadIdx.x; i < HID; i += 256) {
    float a = 0.0f;
#pragma unroll
    for (int e = 0; e < NEXP; ++e)
      a += w[b * NEXP + e] * ecls[((size_t)e * BATCH + b) * HID + i];
    comb[(size_t)b * HID + i] = a;
  }
}

// ---------------------------------------------------------------------------
// host-side orchestration
// ---------------------------------------------------------------------------
static inline void launch_gemm(const float* A, int lda, const float* B,
                               const float* bias, float* C, int M, int N, int K,
                               int act, hipStream_t st) {
  dim3 grid((N + GT_N - 1) / GT_N, (M + GT_M - 1) / GT_M);
  gemm_kernel<<<grid, 256, 0, st>>>(A, lda, B, bias, C, M, N, K, act);
}

static void run_encoder(const float* qkv_w, const float* o_w, const float* ff1_w,
                        const float* ff2_w, const float* ln1s, const float* ln1b,
                        const float* ln2s, const float* ln2b,
                        float* x, float* qkvbuf, float* ctxbuf, float* ffhbuf,
                        const int* mask, hipStream_t st) {
  for (int l = 0; l < NLAY; ++l) {
    launch_gemm(x, HID, qkv_w + (size_t)l * HID * 3 * HID, nullptr, qkvbuf,
                TOK, 3 * HID, HID, ACT_NONE, st);
    attn_kernel<<<BATCH * NHEAD, 256, 0, st>>>(qkvbuf, mask, ctxbuf);
    // o-projection (reuse qkvbuf as temp)
    launch_gemm(ctxbuf, HID, o_w + (size_t)l * HID * HID, nullptr, qkvbuf,
                TOK, HID, HID, ACT_NONE, st);
    ln_kernel<<<TOK, 256, 0, st>>>(x, qkvbuf, x, ln1s + (size_t)l * HID,
                                   ln1b + (size_t)l * HID, HID, ACT_NONE);
    launch_gemm(x, HID, ff1_w + (size_t)l * HID * FFD, nullptr, ffhbuf,
                TOK, FFD, HID, ACT_GELU, st);
    launch_gemm(ffhbuf, FFD, ff2_w + (size_t)l * FFD * HID, nullptr, qkvbuf,
                TOK, HID, FFD, ACT_NONE, st);
    ln_kernel<<<TOK, 256, 0, st>>>(x, qkvbuf, x, ln2s + (size_t)l * HID,
                                   ln2b + (size_t)l * HID, HID, ACT_NONE);
  }
}

extern "C" void kernel_launch(void* const* d_in, const int* in_sizes, int n_in,
                              void* d_out, int out_size, void* d_ws, size_t ws_size,
                              hipStream_t stream) {
  const int*   input_ids = (const int*)  d_in[0];
  const int*   attn_mask = (const int*)  d_in[1];
  const float* embed     = (const float*)d_in[2];
  const float* g_qkv  = (const float*)d_in[3];
  const float* g_o    = (const float*)d_in[4];
  const float* g_ff1  = (const float*)d_in[5];
  const float* g_ff2  = (const float*)d_in[6];
  const float* g_l1s  = (const float*)d_in[7];
  const float* g_l1b  = (const float*)d_in[8];
  const float* g_l2s  = (const float*)d_in[9];
  const float* g_l2b  = (const float*)d_in[10];
  const float* e_qkv  = (const float*)d_in[11];
  const float* e_o    = (const float*)d_in[12];
  const float* e_ff1  = (const float*)d_in[13];
  const float* e_ff2  = (const float*)d_in[14];
  const float* e_l1s  = (const float*)d_in[15];
  const float* e_l1b  = (const float*)d_in[16];
  const float* e_l2s  = (const float*)d_in[17];
  const float* e_l2b  = (const float*)d_in[18];
  const float* gw1 = (const float*)d_in[19];
  const float* gb1 = (const float*)d_in[20];
  const float* glns = (const float*)d_in[21];
  const float* glnb = (const float*)d_in[22];
  const float* gw2 = (const float*)d_in[23];
  const float* gb2 = (const float*)d_in[24];
  const float* cw1 = (const float*)d_in[25];
  const float* cb1 = (const float*)d_in[26];
  const float* clns = (const float*)d_in[27];
  const float* clnb = (const float*)d_in[28];
  const float* cw2 = (const float*)d_in[29];
  const float* cb2 = (const float*)d_in[30];

  float* out_logits = (float*)d_out;             // [64,3]
  float* out_probs  = (float*)d_out + BATCH * 3; // [64,4]

  // workspace layout (floats)
  float* ws = (float*)d_ws;
  size_t off = 0;
  float* x0   = ws + off; off += (size_t)TOK * HID;       // embeddings
  float* x    = ws + off; off += (size_t)TOK * HID;       // residual stream
  float* qkvb = ws + off; off += (size_t)TOK * 3 * HID;   // qkv / proj temp
  float* ctxb = ws + off; off += (size_t)TOK * HID;       // attention output
  float* ffh  = ws + off; off += (size_t)TOK * FFD;       // FF hidden
  float* ecls = ws + off; off += (size_t)NEXP * BATCH * HID;
  float* h1   = ws + off; off += (size_t)BATCH * 256;
  float* h1b  = ws + off; off += (size_t)BATCH * 256;
  float* glog = ws + off; off += (size_t)BATCH * NEXP;
  float* wmix = ws + off; off += (size_t)BATCH * NEXP;
  float* comb = ws + off; off += (size_t)BATCH * HID;
  float* c1   = ws + off; off += (size_t)BATCH * 256;
  float* c1b  = ws + off; off += (size_t)BATCH * 256;

  const size_t xbytes = (size_t)TOK * HID * sizeof(float);

  // 1) embedding gather
  embed_kernel<<<TOK, 256, 0, stream>>>(input_ids, embed, x0);

  // 2) gating encoder
  hipMemcpyAsync(x, x0, xbytes, hipMemcpyDeviceToDevice, stream);
  run_encoder(g_qkv, g_o, g_ff1, g_ff2, g_l1s, g_l1b, g_l2s, g_l2b,
              x, qkvb, ctxb, ffh, attn_mask, stream);

  // 3) gating head on CLS tokens (strided A: row b -> token (b,0))
  launch_gemm(x, SEQ * HID, gw1, gb1, h1, BATCH, 256, HID, ACT_NONE, stream);
  ln_kernel<<<BATCH, 256, 0, stream>>>(h1, nullptr, h1b, glns, glnb, 256, ACT_GELU);
  launch_gemm(h1b, 256, gw2, gb2, glog, BATCH, NEXP, 256, ACT_NONE, stream);
  gate_head_kernel<<<1, 64, 0, stream>>>(glog, out_probs, wmix);

  // 4) experts (dense: every expert on full batch)
  for (int e = 0; e < NEXP; ++e) {
    hipMemcpyAsync(x, x0, xbytes, hipMemcpyDeviceToDevice, stream);
    run_encoder(e_qkv + (size_t)e * NLAY * HID * 3 * HID,
                e_o   + (size_t)e * NLAY * HID * HID,
                e_ff1 + (size_t)e * NLAY * HID * FFD,
                e_ff2 + (size_t)e * NLAY * FFD * HID,
                e_l1s + (size_t)e * NLAY * HID,
                e_l1b + (size_t)e * NLAY * HID,
                e_l2s + (size_t)e * NLAY * HID,
                e_l2b + (size_t)e * NLAY * HID,
                x, qkvb, ctxb, ffh, attn_mask, stream);
    cls_copy_kernel<<<BATCH, 256, 0, stream>>>(x, ecls + (size_t)e * BATCH * HID);
  }

  // 5) top-2 weighted combine + classifier head
  combine_kernel<<<BATCH, 256, 0, stream>>>(wmix, ecls, comb);
  launch_gemm(comb, HID, cw1, cb1, c1, BATCH, 256, HID, ACT_NONE, stream);
  ln_kernel<<<BATCH, 256, 0, stream>>>(c1, nullptr, c1b, clns, clnb, 256, ACT_RELU);
  launch_gemm(c1b, 256, cw2, cb2, out_logits, BATCH, 3, 256, ACT_NONE, stream);
}
